// RCNN_71090298683840
// MI455X (gfx1250) — compile-verified, matrix operand order checked by reading, not run on hardware
//
#include <hip/hip_runtime.h>
#include <hip/hip_bf16.h>

// ---------------------------------------------------------------------------
// Types for CDNA5 WMMA (wave32): V_WMMA_F32_16X16X32_F16
// ---------------------------------------------------------------------------
typedef __attribute__((ext_vector_type(16))) _Float16 v16h;
typedef __attribute__((ext_vector_type(8)))  _Float16 v8h;
typedef __attribute__((ext_vector_type(8)))  float    v8f;

#define BB   256   // batch
#define TT   192   // frames
#define HID  64    // hidden
#define CCH  256   // conv channels / gru steps
#define G3   192   // 3*H
#define KC2  1536  // conv2 contraction = 256*6
#define MBT  49152 // B*T rows
#define MJB  65536 // 256 steps * 256 batch rows
#define OUTC 28

// ---------------------------------------------------------------------------
// Fragment loader for 16x16x32 f16 WMMA (A: 16xK row-major slice; B: N-major,
// K-contiguous).  ISA 7.12.2 16-bit layout: lane L holds row (L&15); element e
// maps to k = k0 + e + (e&8) + ((L&16)>>1).  That is two contiguous runs of 8
// halves -> two 16-byte loads.
// ---------------------------------------------------------------------------
__device__ __forceinline__ v16h load_frag(const _Float16* __restrict__ p,
                                          int row, int ld, int k0, int lane) {
  const int off = (lane & 16) ? 8 : 0;
  const _Float16* q = p + (size_t)row * (size_t)ld + (size_t)(k0 + off);
  v8h lo = *(const v8h*)(q);
  v8h hi = *(const v8h*)(q + 16);
  v16h r;
#pragma unroll
  for (int e = 0; e < 8; ++e) { r[e] = lo[e]; r[e + 8] = hi[e]; }
  return r;
}

__device__ __forceinline__ v8f wmma32(v16h a, v16h b, v8f c) {
  // (neg_a, A, neg_b, B, c_mod, C, reuse_a, reuse_b)
  return __builtin_amdgcn_wmma_f32_16x16x32_f16(false, a, false, b, (short)0, c,
                                                false, false);
}

__device__ __forceinline__ float sigm(float x) { return 1.0f / (1.0f + __expf(-x)); }

// ---------------------------------------------------------------------------
// Weight conversion f32 -> f16 with optional sub-matrix extraction / zero pad.
// dst[r*cols + c] = (r < src_rows) ? src[r*src_ld + c] : 0
// ---------------------------------------------------------------------------
__global__ void cvt_strided(const float* __restrict__ src, _Float16* __restrict__ dst,
                            int rows, int cols, int src_ld, int src_rows) {
  int idx = blockIdx.x * 256 + threadIdx.x;
  int total = rows * cols;
  if (idx >= total) return;
  int r = idx / cols;
  int c = idx - r * cols;
  float v = (r < src_rows) ? src[(size_t)r * src_ld + c] : 0.0f;
  dst[(size_t)r * cols + c] = (_Float16)v;
}

// ---------------------------------------------------------------------------
// Fused conv1 (+ReLU) -> conv2 GEMM.
// Each block owns a 16-row M-tile of (b,t) rows.  Phase 1 builds the
// 16 x 1536 f16 A-tile (conv1 output) in LDS straight from x (no HBM round
// trip).  Phase 2: 4 waves x 4 n-tiles of WMMA against w2h (K-contiguous,
// L2-resident).  Output: featsT[c][b*192+t] = conv2 + bias, stored f16.
// ---------------------------------------------------------------------------
__global__ __launch_bounds__(128) void conv_feats_kernel(
    const float* __restrict__ x, const float* __restrict__ w1,
    const float* __restrict__ b1, const _Float16* __restrict__ w2h,
    const float* __restrict__ b2, _Float16* __restrict__ featsT) {
  __shared__ _Float16 sh_A[16 * 1544] __attribute__((aligned(16)));  // padded ld
  __shared__ float    sh_x[2 * 18 * 63];                             // x patch

  const int tid = threadIdx.x;
  const int mt  = blockIdx.x;
  const int r0  = mt * 16;
  const int b   = r0 / TT;
  const int t0  = r0 - b * TT;

  // ---- stage input patch: v[b, i, t0-1 .. t0+16, 0..20, 0..2] (OOB -> 0)
  for (int idx = tid; idx < 2 * 18 * 63; idx += 128) {
    int i   = idx / (18 * 63);
    int rem = idx - i * (18 * 63);
    int dd  = rem / 63;
    int hw  = rem - dd * 63;
    int d   = t0 + dd - 1;
    float v = 0.0f;
    if (d >= 0 && d < TT)
      v = x[(size_t)b * 24192 + (size_t)i * 12096 + (size_t)d * 63 + hw];
    sh_x[idx] = v;
  }
  __syncthreads();

  // ---- phase 1: conv1 + ReLU -> A-tile (k = c*6 + y)
  for (int idx = tid; idx < 16 * KC2; idx += 128) {
    int m = idx / KC2;
    int k = idx - m * KC2;
    int c = k / 6;
    int y = k - 6 * c;
    float acc = b1[c];
    const float* wc = w1 + c * 72;  // [i][kd][kh][kw]: strides 36,12,3,1
#pragma unroll
    for (int i = 0; i < 2; ++i) {
#pragma unroll
      for (int kd = 0; kd < 3; ++kd) {
        const float* xp  = sh_x + (i * 18 + m + kd) * 63;
        const float* wkd = wc + i * 36 + kd * 12;
#pragma unroll
        for (int kh = 0; kh < 4; ++kh) {
          int hh = 4 * y + kh - 3;
          if (hh >= 0 && hh < 21) {
            acc += xp[hh * 3 + 0] * wkd[kh * 3 + 0];
            acc += xp[hh * 3 + 1] * wkd[kh * 3 + 1];
            acc += xp[hh * 3 + 2] * wkd[kh * 3 + 2];
          }
        }
      }
    }
    sh_A[m * 1544 + k] = (_Float16)fmaxf(acc, 0.0f);
  }
  __syncthreads();

  // ---- phase 2: 16x256 = A(16x1536) * W2(1536x256), WMMA
  const int lane  = tid & 31;
  const int wave  = tid >> 5;
  const int lrow  = lane & 15;
  const int mbase = (lane & 16) ? 8 : 0;
  for (int nt = wave; nt < 16; nt += 4) {
    v8f acc = {};
    for (int k0 = 0; k0 < KC2; k0 += 32) {
      v16h a = load_frag(sh_A, lrow, 1544, k0, lane);
      v16h w = load_frag(w2h, nt * 16 + lrow, KC2, k0, lane);
      acc = wmma32(a, w, acc);
    }
    int n = nt * 16 + lrow;
    float bias = b2[n];
    _Float16 vals[8];
#pragma unroll
    for (int v = 0; v < 8; ++v) vals[v] = (_Float16)(acc[v] + bias);
    // 8 consecutive (b,t) rows at fixed channel n -> one 16B store
    *(v8h*)(featsT + (size_t)n * MBT + (size_t)(r0 + mbase)) = *(const v8h*)vals;
  }
}

// ---------------------------------------------------------------------------
// Generic WMMA GEMM: C(Mx N, f32) = A(f16, row-major, ld=K) * B(f16, N-major,
// K-contiguous) + bias[n].  grid.x = M/16, block = 128 (4 waves).
// Used for GX1 = featsT * Wx1^T + b_ih1  (N=192,K=192)
// and  GX2 = outs1 * W_ih2^T + b_ih2     (N=192,K=64)
// ---------------------------------------------------------------------------
__global__ __launch_bounds__(128) void gemm_f16_kernel(
    const _Float16* __restrict__ A, const _Float16* __restrict__ Bh,
    const float* __restrict__ bias, float* __restrict__ C, int N, int K) {
  const int r0    = blockIdx.x * 16;
  const int tid   = threadIdx.x;
  const int lane  = tid & 31;
  const int wave  = tid >> 5;
  const int lrow  = lane & 15;
  const int mbase = (lane & 16) ? 8 : 0;
  const int ntiles = N >> 4;
  for (int nt = wave; nt < ntiles; nt += 4) {
    v8f acc = {};
    for (int k0 = 0; k0 < K; k0 += 32) {
      v16h a = load_frag(A, r0 + lrow, K, k0, lane);
      v16h w = load_frag(Bh, nt * 16 + lrow, K, k0, lane);
      acc = wmma32(a, w, acc);
    }
    int n = nt * 16 + lrow;
    float bs = bias[n];
#pragma unroll
    for (int v = 0; v < 8; ++v)
      C[(size_t)(r0 + mbase + v) * N + n] = acc[v] + bs;
  }
}

// ---------------------------------------------------------------------------
// GRU1 scan.  grid = 16 blocks (one 16-row batch tile each), block = 128.
// Per step: hx = h*Whx^T, gh = h*Whh^T (both via WMMA, 24 16x16 tiles split
// across 4 waves), then fp32 gates.  GX1 (x-part incl. b_ih1) precomputed.
// ---------------------------------------------------------------------------
__global__ __launch_bounds__(128) void gru1_kernel(
    const float* __restrict__ GX1, const _Float16* __restrict__ whx,
    const _Float16* __restrict__ whh, const float* __restrict__ b_hh,
    _Float16* __restrict__ outs1h) {
  __shared__ float    sh_h[16 * 64];
  __shared__ _Float16 sh_h16[16 * 72] __attribute__((aligned(16)));
  __shared__ float    sh_hx[16 * 196];
  __shared__ float    sh_gh[16 * 196];

  const int tid  = threadIdx.x;
  const int lane = tid & 31;
  const int wave = tid >> 5;
  const int b0   = blockIdx.x * 16;
  const int lrow  = lane & 15;
  const int mbase = (lane & 16) ? 8 : 0;

  for (int idx = tid; idx < 1024; idx += 128) sh_h[idx] = 0.0f;
  __syncthreads();

  for (int j = 0; j < CCH; ++j) {
    // h -> f16 operand tile
    for (int idx = tid; idx < 1024; idx += 128) {
      int m = idx >> 6, hh = idx & 63;
      sh_h16[m * 72 + hh] = (_Float16)sh_h[idx];
    }
    __syncthreads();

    // 2 matmuls x 12 n-tiles = 24 tasks over 4 waves (uniform per wave)
    for (int tt = wave; tt < 24; tt += 4) {
      int mat = tt / 12;
      int nt  = tt - mat * 12;
      const _Float16* W = mat ? whh : whx;
      v8f acc = {};
#pragma unroll
      for (int k0 = 0; k0 < HID; k0 += 32) {
        v16h a = load_frag(sh_h16, lrow, 72, k0, lane);
        v16h w = load_frag(W, nt * 16 + lrow, HID, k0, lane);
        acc = wmma32(a, w, acc);
      }
      float* dst = mat ? sh_gh : sh_hx;
      int n = nt * 16 + lrow;
#pragma unroll
      for (int v = 0; v < 8; ++v) dst[(mbase + v) * 196 + n] = acc[v];
    }
    __syncthreads();

    // gates (fp32)
    for (int idx = tid; idx < 1024; idx += 128) {
      int m = idx >> 6, hh = idx & 63;
      size_t R = (size_t)j * BB + (size_t)(b0 + m);
      const float* gx = GX1 + R * G3;
      float gxr = gx[hh]        + sh_hx[m * 196 + hh];
      float gxz = gx[64 + hh]   + sh_hx[m * 196 + 64 + hh];
      float gxn = gx[128 + hh]  + sh_hx[m * 196 + 128 + hh];
      float ghr = sh_gh[m * 196 + hh]       + b_hh[hh];
      float ghz = sh_gh[m * 196 + 64 + hh]  + b_hh[64 + hh];
      float ghn = sh_gh[m * 196 + 128 + hh] + b_hh[128 + hh];
      float r  = sigm(gxr + ghr);
      float z  = sigm(gxz + ghz);
      float ng = tanhf(gxn + r * ghn);
      float hn = (1.0f - z) * ng + z * sh_h[idx];
      sh_h[idx] = hn;
      outs1h[R * HID + hh] = (_Float16)hn;
    }
    __syncthreads();
  }
}

// ---------------------------------------------------------------------------
// GRU2 scan: input gates fully precomputed in GX2; only gh = h*Whh2^T per
// step.  Writes seq in FC row order: seqh[(b*256 + j)*64 + hh].
// ---------------------------------------------------------------------------
__global__ __launch_bounds__(128) void gru2_kernel(
    const float* __restrict__ GX2, const _Float16* __restrict__ whh,
    const float* __restrict__ b_hh, _Float16* __restrict__ seqh) {
  __shared__ float    sh_h[16 * 64];
  __shared__ _Float16 sh_h16[16 * 72] __attribute__((aligned(16)));
  __shared__ float    sh_gh[16 * 196];

  const int tid  = threadIdx.x;
  const int lane = tid & 31;
  const int wave = tid >> 5;
  const int b0   = blockIdx.x * 16;
  const int lrow  = lane & 15;
  const int mbase = (lane & 16) ? 8 : 0;

  for (int idx = tid; idx < 1024; idx += 128) sh_h[idx] = 0.0f;
  __syncthreads();

  for (int j = 0; j < CCH; ++j) {
    for (int idx = tid; idx < 1024; idx += 128) {
      int m = idx >> 6, hh = idx & 63;
      sh_h16[m * 72 + hh] = (_Float16)sh_h[idx];
    }
    __syncthreads();

    for (int nt = wave; nt < 12; nt += 4) {
      v8f acc = {};
#pragma unroll
      for (int k0 = 0; k0 < HID; k0 += 32) {
        v16h a = load_frag(sh_h16, lrow, 72, k0, lane);
        v16h w = load_frag(whh, nt * 16 + lrow, HID, k0, lane);
        acc = wmma32(a, w, acc);
      }
      int n = nt * 16 + lrow;
#pragma unroll
      for (int v = 0; v < 8; ++v) sh_gh[(mbase + v) * 196 + n] = acc[v];
    }
    __syncthreads();

    for (int idx = tid; idx < 1024; idx += 128) {
      int m = idx >> 6, hh = idx & 63;
      size_t R = (size_t)j * BB + (size_t)(b0 + m);
      const float* gx = GX2 + R * G3;  // includes b_ih2
      float ghr = sh_gh[m * 196 + hh]       + b_hh[hh];
      float ghz = sh_gh[m * 196 + 64 + hh]  + b_hh[64 + hh];
      float ghn = sh_gh[m * 196 + 128 + hh] + b_hh[128 + hh];
      float r  = sigm(gx[hh] + ghr);
      float z  = sigm(gx[64 + hh] + ghz);
      float ng = tanhf(gx[128 + hh] + r * ghn);
      float hn = (1.0f - z) * ng + z * sh_h[idx];
      sh_h[idx] = hn;
      seqh[((size_t)(b0 + m) * CCH + (size_t)j) * HID + hh] = (_Float16)hn;
    }
    __syncthreads();
  }
}

// ---------------------------------------------------------------------------
// Fused FC1(relu) + FC2.  One wave per 16-row M-tile of seq (65536 rows).
// fc2h is zero-padded to 32 rows so the second stage is two clean n-tiles.
// ---------------------------------------------------------------------------
__global__ __launch_bounds__(32) void fc_kernel(
    const _Float16* __restrict__ seqh, const _Float16* __restrict__ fc1h,
    const float* __restrict__ fc1_b, const _Float16* __restrict__ fc2h,
    const float* __restrict__ fc2_b, float* __restrict__ out) {
  __shared__ _Float16 mid[16 * 72] __attribute__((aligned(16)));
  const int r0    = blockIdx.x * 16;
  const int lane  = threadIdx.x & 31;
  const int lrow  = lane & 15;
  const int mbase = (lane & 16) ? 8 : 0;

#pragma unroll
  for (int nt = 0; nt < 4; ++nt) {
    v8f acc = {};
#pragma unroll
    for (int k0 = 0; k0 < HID; k0 += 32) {
      v16h a = load_frag(seqh, r0 + lrow, HID, k0, lane);
      v16h w = load_frag(fc1h, nt * 16 + lrow, HID, k0, lane);
      acc = wmma32(a, w, acc);
    }
    int n = nt * 16 + lrow;
    float bs = fc1_b[n];
#pragma unroll
    for (int v = 0; v < 8; ++v)
      mid[(mbase + v) * 72 + n] = (_Float16)fmaxf(acc[v] + bs, 0.0f);
  }
  __syncthreads();

#pragma unroll
  for (int nt = 0; nt < 2; ++nt) {
    v8f acc = {};
#pragma unroll
    for (int k0 = 0; k0 < HID; k0 += 32) {
      v16h a = load_frag(mid, lrow, 72, k0, lane);
      v16h w = load_frag(fc2h, nt * 16 + lrow, HID, k0, lane);
      acc = wmma32(a, w, acc);
    }
    int n = nt * 16 + lrow;
    if (n < OUTC) {
      float bs = fc2_b[n];
#pragma unroll
      for (int v = 0; v < 8; ++v)
        out[(size_t)(r0 + mbase + v) * OUTC + n] = acc[v] + bs;
    }
  }
}

// ---------------------------------------------------------------------------
// Host launcher
// ---------------------------------------------------------------------------
extern "C" void kernel_launch(void* const* d_in, const int* in_sizes, int n_in,
                              void* d_out, int out_size, void* d_ws, size_t ws_size,
                              hipStream_t stream) {
  (void)in_sizes; (void)n_in; (void)out_size; (void)ws_size;
  const float* x       = (const float*)d_in[0];
  const float* conv1_w = (const float*)d_in[1];
  const float* conv1_b = (const float*)d_in[2];
  const float* conv2_w = (const float*)d_in[3];
  const float* conv2_b = (const float*)d_in[4];
  const float* w_ih1   = (const float*)d_in[5];
  const float* w_hh1   = (const float*)d_in[6];
  const float* b_ih1   = (const float*)d_in[7];
  const float* b_hh1   = (const float*)d_in[8];
  const float* w_ih2   = (const float*)d_in[9];
  const float* w_hh2   = (const float*)d_in[10];
  const float* b_ih2   = (const float*)d_in[11];
  const float* b_hh2   = (const float*)d_in[12];
  const float* fc1_w   = (const float*)d_in[13];
  const float* fc1_b   = (const float*)d_in[14];
  const float* fc2_w   = (const float*)d_in[15];
  const float* fc2_b   = (const float*)d_in[16];
  float* out = (float*)d_out;

  // --- carve workspace (256B aligned regions), ~93 MB total
  size_t off = 0;
  auto carve = [&](size_t bytes) -> void* {
    void* p = (char*)d_ws + off;
    off += (bytes + 255) & ~(size_t)255;
    return p;
  };
  _Float16* featsT = (_Float16*)carve((size_t)CCH * MBT * 2);   // 25.2 MB
  float*    GX     = (float*)carve((size_t)MJB * G3 * 4);       // 50.3 MB (GX1 then GX2)
  _Float16* outs1h = (_Float16*)carve((size_t)MJB * HID * 2);   // 8.4 MB
  _Float16* seqh   = (_Float16*)carve((size_t)MJB * HID * 2);   // 8.4 MB
  _Float16* w2h    = (_Float16*)carve((size_t)CCH * KC2 * 2);   // 0.79 MB
  _Float16* wx1h   = (_Float16*)carve((size_t)G3 * TT * 2);
  _Float16* whx1h  = (_Float16*)carve((size_t)G3 * HID * 2);
  _Float16* whh1h  = (_Float16*)carve((size_t)G3 * HID * 2);
  _Float16* wx2h   = (_Float16*)carve((size_t)G3 * HID * 2);
  _Float16* whh2h  = (_Float16*)carve((size_t)G3 * HID * 2);
  _Float16* fc1h   = (_Float16*)carve((size_t)64 * HID * 2);
  _Float16* fc2h   = (_Float16*)carve((size_t)32 * HID * 2);

  auto cvt = [&](const float* src, _Float16* dst, int rows, int cols, int ld, int srows) {
    int total = rows * cols;
    cvt_strided<<<(total + 255) / 256, 256, 0, stream>>>(src, dst, rows, cols, ld, srows);
  };
  // weight conversions (f32 -> f16, WMMA B-operand layouts: N-major, K-contig)
  cvt(conv2_w,      w2h,   CCH, KC2, KC2, CCH);  // already [c_out][c_in*6+y]
  cvt(w_ih1,        wx1h,  G3,  TT,  CCH, G3);   // x-columns of W_ih1
  cvt(w_ih1 + TT,   whx1h, G3,  HID, CCH, G3);   // h-columns of W_ih1
  cvt(w_hh1,        whh1h, G3,  HID, HID, G3);
  cvt(w_ih2,        wx2h,  G3,  HID, HID, G3);
  cvt(w_hh2,        whh2h, G3,  HID, HID, G3);
  cvt(fc1_w,        fc1h,  64,  HID, HID, 64);
  cvt(fc2_w,        fc2h,  32,  HID, HID, OUTC); // rows 28..31 zero-padded

  // 1) fused conv1+conv2 -> featsT (f16, [c][b*192+t])
  conv_feats_kernel<<<MBT / 16, 128, 0, stream>>>(x, conv1_w, conv1_b, w2h,
                                                  conv2_b, featsT);

  // 2) GX1 = featsT(65536x192) * Wx1^T + b_ih1   (rows are exactly j*256+b)
  gemm_f16_kernel<<<MJB / 16, 128, 0, stream>>>(featsT, wx1h, b_ih1, GX, G3, TT);

  // 3) GRU1 scan (batch-parallel over 16 workgroups)
  gru1_kernel<<<16, 128, 0, stream>>>(GX, whx1h, whh1h, b_hh1, outs1h);

  // 4) GX2 = outs1(65536x64) * W_ih2^T + b_ih2   (reuses GX buffer)
  gemm_f16_kernel<<<MJB / 16, 128, 0, stream>>>(outs1h, wx2h, b_ih2, GX, G3, HID);

  // 5) GRU2 scan -> seq in (b, j) row order
  gru2_kernel<<<16, 128, 0, stream>>>(GX, whh2h, b_hh2, seqh);

  // 6) fused FC1(relu)+FC2 -> out (B,256,28)
  fc_kernel<<<MJB / 16, 32, 0, stream>>>(seqh, fc1h, fc1_b, fc2h, fc2_b, out);
}
